// Model_39676907887597
// MI455X (gfx1250) — matrix operand order repaired and checked by reading, not compiled, not run on hardware
//
#include <hip/hip_runtime.h>

#define B_ 4
#define S_ 4096
#define D_ 64
#define QT_ (S_/16)      // 256 query tiles per batch
#define KT16_ (S_/16)    // 256 key tiles of 16
#define KT32_ (S_/32)    // 128 key tiles of 32
#define SCALE_INV (1.0f / 140737488355328.0f)   // 2^-47, exact in f32

typedef __attribute__((ext_vector_type(16))) _Float16 v16h;
typedef __attribute__((ext_vector_type(8)))  _Float16 v8h;
typedef __attribute__((ext_vector_type(2)))  _Float16 h2;
typedef __attribute__((ext_vector_type(8)))  float    v8f;

// ---------------------------------------------------------------------------
// Kernel 1: q/k projections + v copy, f32 -> f16, written directly in WMMA
// fragment layouts (own contiguous k order, same for A and B, not the ISA table order) so the flash kernel loads are
// contiguous 32B-per-lane b128s.
//
//  Q (A-layout, 16x32 f16): lane = h*16 + M  (h = lanehalf),
//      half t -> K = h*16 + t
//  K,V (B-layout, 32x16 f16): lane = h*16 + N, half t -> K = h*16 + t
// ---------------------------------------------------------------------------
__global__ void __launch_bounds__(256)
proj_kernel(const float* __restrict__ v1,
            const float* __restrict__ kw, const float* __restrict__ kbias,
            const float* __restrict__ qw, const float* __restrict__ qbias,
            _Float16* __restrict__ qf, _Float16* __restrict__ kf,
            _Float16* __restrict__ vf) {
    const int n   = B_ * S_ * D_;
    int tid = blockIdx.x * blockDim.x + threadIdx.x;   // 0 .. 2n-1
    const bool isK = tid >= n;
    int id = isK ? tid - n : tid;
    int e = id & (D_ - 1);
    int s = (id >> 6) & (S_ - 1);
    int b = id >> 18;                                   // id / (S_*D_)

    const float* w    = isK ? kw : qw;
    const float* bias = isK ? kbias : qbias;
    const float* row  = v1 + ((size_t)b * S_ + s) * D_;

    float acc = bias[e];
    #pragma unroll 8
    for (int d = 0; d < D_; ++d) acc += row[d] * w[e * D_ + d];
    _Float16 h = (_Float16)acc;

    int chunk = e >> 5, ec = e & 31;
    if (!isK) {
        // ---- q: A-fragment layout
        int qt = s >> 4, M = s & 15;
        int hg = ec >> 4;
        int t  = ec & 15;
        int lane = hg * 16 + M;
        qf[((((size_t)b * QT_ + qt) * 2 + chunk) * 32 + lane) * 16 + t] = h;
        // ---- v passthrough: B-fragment layout (K-dim = key index within 32)
        int kt32 = s >> 5, K = s & 31;
        int nt = e >> 4, N = e & 15;
        int vlane = (K >> 4) * 16 + N;
        vf[((((size_t)b * KT32_ + kt32) * 4 + nt) * 32 + vlane) * 16 + (K & 15)] =
            (_Float16)row[e];
    } else {
        // ---- k: B-fragment layout (K-dim = d, N = key index within 16)
        int kt = s >> 4, N = s & 15;
        int hg = ec >> 4;
        int t  = ec & 15;
        int lane = hg * 16 + N;
        kf[((((size_t)b * KT16_ + kt) * 2 + chunk) * 32 + lane) * 16 + t] = h;
    }
}

// ---------------------------------------------------------------------------
// Kernel 2: flash attention. 1 wave == one 16-query tile; 32 keys per
// iteration; 9 v_wmma_f32_16x16x32_f16 per iteration (4 QK^T, 4 P*V, and one
// P*Ones that tracks the softmax denominator l in matrix hardware).
// Row-max reduction runs on packed f16 pairs (v_pk_max + 32-bit bpermute).
// ---------------------------------------------------------------------------
#define WMMA16(A, Bm, C) \
    __builtin_amdgcn_wmma_f32_16x16x32_f16(false, (A), false, (Bm), (short)0, (C), false, false)

__global__ void __launch_bounds__(128)
flash_kernel(const _Float16* __restrict__ qf, const _Float16* __restrict__ kf,
             const _Float16* __restrict__ vf, float* __restrict__ out) {
    __shared__ _Float16 plds[4][16 * 32];    // one 1KB P-tile staging area per wave

    const int wave = threadIdx.x >> 5;
    const int lane = threadIdx.x & 31;
    const int gw = blockIdx.x * 4 + wave;    // global query-tile index
    const int b  = gw >> 8;                  // QT_ = 256 tiles per batch
    const int qt = gw & 255;
    const int hg = lane >> 4;                // lane half: selects row group M = hg*8+r
    const int nn = lane & 15;

    // Q fragments for this tile (both d-chunks), resident all kernel
    const _Float16* qb = qf + (((size_t)b * QT_ + qt) * 2) * 512 + (size_t)lane * 16;
    const v16h qa0 = *(const v16h*)(qb);
    const v16h qa1 = *(const v16h*)(qb + 512);

    // all-ones B fragment: o4 = sum_k P[q,k] in every column
    v16h ones;
    #pragma unroll
    for (int t = 0; t < 16; ++t) ones[t] = (_Float16)1.0f;

    v8f o0 = {}, o1 = {}, o2 = {}, o3 = {};   // 16x64 f32 output accumulators
    v8f o4 = {};                              // softmax denominator accumulator
    float m[8];                               // running row max (unscaled units)
    #pragma unroll
    for (int r = 0; r < 8; ++r) m[r] = -1e30f;

    _Float16* pl = plds[wave];

    for (int kt = 0; kt < KT32_; ++kt) {
        // ---------------- S = Q * K^T  (two 16x16 N-subtiles) ----------------
        const _Float16* kb0 = kf + (((size_t)b * KT16_ + (kt * 2 + 0)) * 2) * 512 + (size_t)lane * 16;
        const _Float16* kb1 = kf + (((size_t)b * KT16_ + (kt * 2 + 1)) * 2) * 512 + (size_t)lane * 16;
        v16h kb00 = *(const v16h*)(kb0);
        v16h kb01 = *(const v16h*)(kb0 + 512);
        v16h kb10 = *(const v16h*)(kb1);
        v16h kb11 = *(const v16h*)(kb1 + 512);

        v8f s0 = {}, s1 = {};
        s0 = WMMA16(qa0, kb00, s0);
        s0 = WMMA16(qa1, kb01, s0);
        s1 = WMMA16(qa0, kb10, s1);
        s1 = WMMA16(qa1, kb11, s1);

        if (kt + 1 < KT32_) {   // prefetch next tile's K/V toward near caches
            __builtin_prefetch(kf + (((size_t)b * KT16_ + (kt + 1) * 2) * 2) * 512 + (size_t)lane * 16, 0, 3);
            __builtin_prefetch(vf + ((((size_t)b * KT32_ + (kt + 1)) * 4) * 32 + lane) * 16, 0, 3);
        }

        // ------- online softmax: row max, packed-f16 cross-lane reduction ----
        // any row-uniform m is a valid softmax shift; f16 precision suffices
        h2 pm[4];
        #pragma unroll
        for (int r = 0; r < 4; ++r) {
            h2 a;
            a[0] = (_Float16)fmaxf(s0[2 * r],     s1[2 * r]);
            a[1] = (_Float16)fmaxf(s0[2 * r + 1], s1[2 * r + 1]);
            pm[r] = a;
        }
        #pragma unroll
        for (int off = 1; off < 16; off <<= 1) {
            #pragma unroll
            for (int r = 0; r < 4; ++r) {
                int sh = __shfl_xor(__builtin_bit_cast(int, pm[r]), off, 32);
                pm[r] = __builtin_elementwise_max(pm[r], __builtin_bit_cast(h2, sh));
            }
        }
        float alpha[8], mc[8];
        #pragma unroll
        for (int r = 0; r < 8; ++r) {
            float rmax = (float)pm[r >> 1][r & 1];
            float mn = fmaxf(m[r], rmax);
            alpha[r] = __expf((m[r] - mn) * SCALE_INV);
            m[r] = mn;
            mc[r] = mn * SCALE_INV;
        }
        // P = exp(s*2^-47 - m*2^-47); write straight to LDS in f16
        #pragma unroll
        for (int r = 0; r < 8; ++r) {
            float p0 = __expf(fmaf(s0[r], SCALE_INV, -mc[r]));
            float p1 = __expf(fmaf(s1[r], SCALE_INV, -mc[r]));
            int row = hg * 8 + r;
            pl[row * 32 + nn]      = (_Float16)p0;
            pl[row * 32 + 16 + nn] = (_Float16)p1;
        }
        // rescale accumulators (including denominator) by alpha
        #pragma unroll
        for (int r = 0; r < 8; ++r) {
            o0[r] *= alpha[r]; o1[r] *= alpha[r];
            o2[r] *= alpha[r]; o3[r] *= alpha[r];
            o4[r] *= alpha[r];
        }

        asm volatile("s_wait_dscnt 0" ::: "memory");
        // A-layout reload: lane reads row (lane&15), cols {hg*16..+15}
        v8h pa_lo = *(const v8h*)(pl + (lane & 15) * 32 + hg * 16);
        v8h pa_hi = *(const v8h*)(pl + (lane & 15) * 32 + hg * 16 + 8);
        v16h pa = __builtin_shufflevector(pa_lo, pa_hi,
                                          0, 1, 2, 3, 4, 5, 6, 7,
                                          8, 9, 10, 11, 12, 13, 14, 15);

        // ---------------- O += P * V ; l += P * Ones --------------------------
        const _Float16* vb = vf + ((((size_t)b * KT32_ + kt) * 4) * 32 + lane) * 16;
        v16h vb0 = *(const v16h*)(vb);
        v16h vb1 = *(const v16h*)(vb + 512);
        v16h vb2 = *(const v16h*)(vb + 1024);
        v16h vb3 = *(const v16h*)(vb + 1536);
        o0 = WMMA16(pa, vb0, o0);
        o1 = WMMA16(pa, vb1, o1);
        o2 = WMMA16(pa, vb2, o2);
        o3 = WMMA16(pa, vb3, o3);
        o4 = WMMA16(pa, ones, o4);
    }

    // ---------------- epilogue: O / l (fast rcp), f32 store ------------------
    float* ob = out + ((size_t)b * S_ + qt * 16) * D_;
    #pragma unroll
    for (int r = 0; r < 8; ++r) {
        int row = hg * 8 + r;
        float inv = __builtin_amdgcn_rcpf(o4[r]);
        ob[row * D_ +  0 + nn] = o0[r] * inv;
        ob[row * D_ + 16 + nn] = o1[r] * inv;
        ob[row * D_ + 32 + nn] = o2[r] * inv;
        ob[row * D_ + 48 + nn] = o3[r] * inv;
    }
}

extern "C" void kernel_launch(void* const* d_in, const int* in_sizes, int n_in,
                              void* d_out, int out_size, void* d_ws, size_t ws_size,
                              hipStream_t stream) {
    (void)in_sizes; (void)n_in; (void)out_size; (void)ws_size;
    const float* v1 = (const float*)d_in[0];
    const float* kw = (const float*)d_in[1];
    const float* kb = (const float*)d_in[2];
    const float* qw = (const float*)d_in[3];
    const float* qb = (const float*)d_in[4];
    float* out = (float*)d_out;

    const size_t nElem = (size_t)B_ * S_ * D_;           // 1,048,576
    _Float16* qf = (_Float16*)d_ws;                      // 2 MB
    _Float16* kf = qf + nElem;                           // 2 MB
    _Float16* vf = kf + nElem;                           // 2 MB (6 MB total ws)

    proj_kernel<<<(2 * (int)nElem) / 256, 256, 0, stream>>>(v1, kw, kb, qw, qb, qf, kf, vf);
    flash_kernel<<<(B_ * QT_) / 4, 128, 0, stream>>>(qf, kf, vf, out);
}
